// MultiHeadedAttention_14096082666376
// MI455X (gfx1250) — matrix operand order repaired and checked by reading, not compiled, not run on hardware
//
#include <hip/hip_runtime.h>
#include <hip/hip_bf16.h>

// ---------------------------------------------------------------------------
// MHA forward for MI455X (gfx1250, wave32, WMMA + Tensor Data Mover).
//   ~137 GFLOP / ~220 MB fused traffic -> f16 intermediates, flash attention,
//   v_wmma_f32_16x16x32_f16 for every matmul, TDM async tile loads for the
//   f16-resident tiles (Q/K tiles in attention, X tile in final projection).
// Pipeline: 3x (X*W^T+b -> f16 [B,H,S,hd] in ws) -> flash attn (f16 ctx in
// [B,S,D] in ws) -> ctx*Wo^T+bo -> f32 d_out.
// ---------------------------------------------------------------------------

typedef __attribute__((ext_vector_type(16))) _Float16 v16h;
typedef __attribute__((ext_vector_type(8)))  _Float16 v8h;
typedef __attribute__((ext_vector_type(8)))  float    v8f;
typedef __attribute__((ext_vector_type(4)))  unsigned int v4u;
typedef __attribute__((ext_vector_type(8)))  int      v8i;
typedef __attribute__((ext_vector_type(4)))  int      v4i;

#define D_MODEL  1024
#define N_HEADS  16
#define HEAD_DIM 64
#define BATCH    4
#define SEQ      2048

// --- WMMA fragment loaders (contiguous k order, same for A and B) --------------
// A (16x32 f16): lane = (m = lane&15, half = lane>>4); element e holds
//   k = 16*half + e  -> one contiguous 16-half (32B) run along k.
__device__ __forceinline__ v16h load_a_frag(const _Float16* row, int kbase, int half) {
  return *(const v16h*)(row + kbase + half * 16);
}
// B (32x16 f16): lane = (n = lane&15, half = lane>>4); element e holds
//   k = 16*half + e  -> one contiguous 16-half (32B) run along k.
__device__ __forceinline__ v16h load_b_frag(const _Float16* row, int kbase, int half) {
  return *(const v16h*)(row + kbase + half * 16);
}
__device__ __forceinline__ v8f wmma16(v16h a, v16h b, v8f c) {
  return __builtin_amdgcn_wmma_f32_16x16x32_f16(false, a, false, b, (short)0, c,
                                                false, false);
}

// --- Tensor Data Mover: async 2D f16 tile load global->LDS (ISA ch.8 D#) ----
// Loads tile_h rows x tile_w f16 from a row-major tensor (row stride
// row_stride_elems), inserting LDS row padding via the TDM pad unit:
//   pad_interval_code: pad every (2<<code) DWORDs; pad_amount_code: (code+1)
//   DWORDs of pad. Issue from ONE wave; wait with s_wait_tensorcnt.
__device__ __forceinline__ void tdm_load_tile_f16(unsigned int lds_byte_off,
                                                  const void* gsrc,
                                                  unsigned int tile_w,
                                                  unsigned int tile_h,
                                                  unsigned int tensor_w,
                                                  unsigned int tensor_h,
                                                  unsigned int row_stride_elems,
                                                  unsigned int pad_interval_code,
                                                  unsigned int pad_amount_code) {
  const unsigned long long ga = (unsigned long long)(uintptr_t)gsrc;
  v4u g0;
  g0[0] = 1u;                                        // count=1, user descriptor
  g0[1] = lds_byte_off;                              // lds_addr
  g0[2] = (unsigned int)(ga & 0xFFFFFFFFu);          // global_addr[31:0]
  g0[3] = (unsigned int)((ga >> 32) & 0x01FFFFFFu)   // global_addr[56:32]
          | (2u << 30);                              // type = 2 ("image")
  v8i g1;
  g1[0] = (int)((1u << 16)                           // data_size = 2 bytes
                | (1u << 20)                         // pad_enable
                | (pad_interval_code << 22) | (pad_amount_code << 25));
  g1[1] = (int)((tensor_w & 0xFFFFu) << 16);         // tensor_dim0[15:0]
  g1[2] = (int)(((tensor_w >> 16) & 0xFFFFu) | ((tensor_h & 0xFFFFu) << 16));
  g1[3] = (int)(((tensor_h >> 16) & 0xFFFFu) | ((tile_w & 0xFFFFu) << 16));
  g1[4] = (int)(tile_h & 0xFFFFu);                   // tile_dim1; tile_dim2=0
  g1[5] = (int)row_stride_elems;                     // tensor_dim0_stride[31:0]
  g1[6] = 0;
  g1[7] = 0;
  const v4i gz = {0, 0, 0, 0};
#if __clang_major__ >= 23
  const v8i gz8 = {0, 0, 0, 0, 0, 0, 0, 0};
  __builtin_amdgcn_tensor_load_to_lds(g0, g1, gz, gz, gz8, 0);
#else
  __builtin_amdgcn_tensor_load_to_lds(g0, g1, gz, gz, 0);
#endif
}

__device__ __forceinline__ unsigned int lds_offset_of(const void* p) {
  // generic -> LDS addrspace cast yields the 32-bit LDS byte offset
  return (unsigned int)(uintptr_t)(__attribute__((address_space(3))) const void*)p;
}

// ---------------------------------------------------------------------------
// GEMM: Y[M,N] = X[M,K] * W[N,K]^T + bias.  64x128 block tile, 8 waves,
// each wave a 32x32 tile (2x2 WMMA), BK=32 (one wmma per tile per k-step).
// HEAD_OUT=true  -> f16 output scattered to [B,H,S,hd]
// HEAD_OUT=false -> f32 output row-major [M,N] (final projection)
// f16 input (final projection): X tile staged by the TDM with pad->stride 48.
// ---------------------------------------------------------------------------
template <typename TIn, bool HEAD_OUT>
__global__ __launch_bounds__(256)
void gemm_wmma_kernel(const TIn* __restrict__ X, const float* __restrict__ W,
                      const float* __restrict__ bias, void* __restrict__ outp,
                      int M, int N, int K) {
  constexpr int BM = 64, BN = 128, BK = 32, LSTR = 48;  // 96B rows: 16/32B aligned
  constexpr bool TDM_X = (sizeof(TIn) == 2);            // f16 input -> TDM staging
  __shared__ __align__(32) _Float16 Xs[BM * LSTR];
  __shared__ __align__(32) _Float16 Ws[BN * LSTR];

  const int tid = threadIdx.x;
  const int lane = tid & 31, wid = tid >> 5;
  const int half = lane >> 4, ln = lane & 15;
  const int m0 = blockIdx.y * BM, n0 = blockIdx.x * BN;
  const int wm = (wid & 1) * 32, wn = (wid >> 1) * 32;

  v8f acc[2][2] = {};

  const int xr = tid >> 2, xc = (tid & 3) * 8;    // 64x32: 8 elems/thread
  const int wr = tid >> 1, wc = (tid & 1) * 16;   // 128x32: 16 elems/thread
  const unsigned int xs_lds = lds_offset_of(Xs);

  for (int k0 = 0; k0 < K; k0 += BK) {
    if (TDM_X) {
      if (wid == 0)  // 64x32 f16 tile; pad every 16 DWORDs (+8) -> stride 48
        tdm_load_tile_f16(xs_lds, X + (size_t)m0 * K + k0, BK, BM, BK, BM,
                          (unsigned int)K, /*interval 16dw*/ 3, /*pad 8dw*/ 7);
    } else {  // stage X tile with f32->f16 conversion
      const TIn* src = X + (size_t)(m0 + xr) * K + k0 + xc;
      v8h t;
#pragma unroll
      for (int i = 0; i < 8; ++i) t[i] = (_Float16)src[i];
      *(v8h*)(Xs + xr * LSTR + xc) = t;
    }
    {  // stage W tile row-major [n][k] (B fragment reads are contiguous)
      const float* src = W + (size_t)(n0 + wr) * K + k0 + wc;
      v8h t0, t1;
#pragma unroll
      for (int i = 0; i < 8; ++i) { t0[i] = (_Float16)src[i]; t1[i] = (_Float16)src[8 + i]; }
      *(v8h*)(Ws + wr * LSTR + wc) = t0;
      *(v8h*)(Ws + wr * LSTR + wc + 8) = t1;
    }
    if (k0 + BK < K) {  // global_prefetch_b8 of next k tiles
      __builtin_prefetch(X + (size_t)(m0 + xr) * K + k0 + BK + xc, 0, 1);
      __builtin_prefetch(W + (size_t)(n0 + wr) * K + k0 + BK + wc, 0, 1);
    }
    if (TDM_X && wid == 0) __builtin_amdgcn_s_wait_tensorcnt(0);
    __syncthreads();
    const v16h a0 = load_a_frag(Xs + (wm + ln) * LSTR, 0, half);
    const v16h a1 = load_a_frag(Xs + (wm + 16 + ln) * LSTR, 0, half);
    const v16h b0 = load_b_frag(Ws + (wn + ln) * LSTR, 0, half);
    const v16h b1 = load_b_frag(Ws + (wn + 16 + ln) * LSTR, 0, half);
    acc[0][0] = wmma16(a0, b0, acc[0][0]);
    acc[0][1] = wmma16(a0, b1, acc[0][1]);
    acc[1][0] = wmma16(a1, b0, acc[1][0]);
    acc[1][1] = wmma16(a1, b1, acc[1][1]);
    __syncthreads();
  }

  // Epilogue. C layout: element v <-> row m = v + 8*half, lane&15 <-> col n.
#pragma unroll
  for (int ni = 0; ni < 2; ++ni) {
    const int n = n0 + wn + ni * 16 + ln;
    const float bb = bias[n];
#pragma unroll
    for (int mi = 0; mi < 2; ++mi) {
#pragma unroll
      for (int vv = 0; vv < 8; ++vv) {
        const int m = m0 + wm + mi * 16 + vv + 8 * half;
        const float val = acc[mi][ni][vv] + bb;
        if (HEAD_OUT) {
          _Float16* out = (_Float16*)outp;
          const int bi = m / SEQ, s = m % SEQ;
          const int h = n / HEAD_DIM, d = n % HEAD_DIM;
          out[(((size_t)bi * N_HEADS + h) * SEQ + s) * HEAD_DIM + d] = (_Float16)val;
        } else {
          ((float*)outp)[(size_t)m * N + n] = val;
        }
      }
    }
  }
}

// ---------------------------------------------------------------------------
// Flash attention: grid = (S/64, B*H); 128 threads = 4 waves, each wave owns
// 16 query rows x full HEAD_DIM. Q/K 64x64 f16 tiles are loaded by the TDM
// (pad 32dw->+8dw reproduces the padded LDS stride 80); V is staged
// transposed manually (TDM cannot transpose). Per 64-key block: scores via
// WMMA, online softmax (1 thread/row), rescale f32 accumulators, ctx += P*V.
// ---------------------------------------------------------------------------
__global__ __launch_bounds__(128)
void flash_attn_kernel(const _Float16* __restrict__ q, const _Float16* __restrict__ k,
                       const _Float16* __restrict__ v,
                       const unsigned char* __restrict__ mask,
                       _Float16* __restrict__ ctx) {
  constexpr int LQ = 80;  // 160B rows: 16B/32B aligned, bank-spread
  __shared__ __align__(32) _Float16 Qs[64 * LQ];
  __shared__ __align__(32) _Float16 Ks[64 * LQ];
  __shared__ __align__(32) _Float16 Vts[64 * LQ];   // V^T: [d][j]
  __shared__ float Ss[64 * 65];                     // f32 scores, padded
  __shared__ __align__(32) _Float16 Ps[64 * LQ];    // exp(scores) f16
  __shared__ float alpha_s[64];
  __shared__ float scale_s[64];

  const int tid = threadIdx.x;
  const int lane = tid & 31, wid = tid >> 5;
  const int half = lane >> 4, ln = lane & 15;
  const int bh = blockIdx.y;
  const int bi = bh / N_HEADS, h = bh % N_HEADS;
  const int q0 = blockIdx.x * 64;

  const _Float16* Qp = q + ((size_t)bh * SEQ + q0) * HEAD_DIM;
  const _Float16* Kp = k + (size_t)bh * SEQ * HEAD_DIM;
  const _Float16* Vp = v + (size_t)bh * SEQ * HEAD_DIM;
  const unsigned char* Mp = mask + (size_t)bi * SEQ;

  // TDM: stage Q block [64 x 64] with pad (32dw interval, +8dw) -> stride 80
  if (wid == 0) {
    tdm_load_tile_f16(lds_offset_of(Qs), Qp, HEAD_DIM, 64, HEAD_DIM, 64,
                      HEAD_DIM, /*interval 32dw*/ 4, /*pad 8dw*/ 7);
    __builtin_amdgcn_s_wait_tensorcnt(0);
  }

  const int qr = wid * 16;          // this wave's query rows
  v8f o[4] = {};                    // 16 x 64 ctx accumulator
  float m_r = -1e30f, l_r = 0.0f;   // running max/sum (valid for tid<64)

  for (int kb = 0; kb < SEQ / 64; ++kb) {
    // TDM: K block row-major into padded LDS
    if (wid == 0)
      tdm_load_tile_f16(lds_offset_of(Ks), Kp + (size_t)kb * 64 * HEAD_DIM,
                        HEAD_DIM, 64, HEAD_DIM, 64, HEAD_DIM, 4, 7);
    {  // stage V transposed (scatter; TDM cannot transpose)
      const int r = tid >> 1, c = (tid & 1) * 32;
      const _Float16* vs = Vp + (size_t)(kb * 64 + r) * HEAD_DIM + c;
#pragma unroll
      for (int i = 0; i < 32; ++i) Vts[(c + i) * LQ + r] = vs[i];
    }
    if (wid == 0) __builtin_amdgcn_s_wait_tensorcnt(0);
    __syncthreads();

    // scores S = (Q K^T) / sqrt(hd): 16x64 per wave, k-dim 64 = 2 wmma steps
#pragma unroll
    for (int jt = 0; jt < 4; ++jt) {
      v8f s = {};
      s = wmma16(load_a_frag(Qs + (qr + ln) * LQ, 0, half),
                 load_b_frag(Ks + (jt * 16 + ln) * LQ, 0, half), s);
      s = wmma16(load_a_frag(Qs + (qr + ln) * LQ, 32, half),
                 load_b_frag(Ks + (jt * 16 + ln) * LQ, 32, half), s);
#pragma unroll
      for (int vv = 0; vv < 8; ++vv)
        Ss[(qr + vv + 8 * half) * 65 + jt * 16 + ln] = s[vv] * 0.125f;
    }
    __syncthreads();

    // online softmax: one row per thread (tid < 64)
    if (tid < 64) {
      const float* srow = Ss + tid * 65;
      float mnew = m_r;
#pragma unroll 8
      for (int j = 0; j < 64; ++j) {
        const float x = Mp[kb * 64 + j] ? srow[j] : -1e30f;
        mnew = fmaxf(mnew, x);
      }
      const float alpha = __expf(m_r - mnew);
      float lsum = 0.0f;
#pragma unroll 8
      for (int j = 0; j < 64; ++j) {
        const float x = Mp[kb * 64 + j] ? srow[j] : -1e30f;
        const float p = __expf(x - mnew);
        lsum += p;
        Ps[tid * LQ + j] = (_Float16)p;
      }
      alpha_s[tid] = alpha;
      l_r = l_r * alpha + lsum;
      m_r = mnew;
    }
    __syncthreads();

    // rescale accumulator, then ctx += P * V
#pragma unroll
    for (int nt = 0; nt < 4; ++nt) {
#pragma unroll
      for (int vv = 0; vv < 8; ++vv) o[nt][vv] *= alpha_s[qr + vv + 8 * half];
      o[nt] = wmma16(load_a_frag(Ps + (qr + ln) * LQ, 0, half),
                     load_b_frag(Vts + (nt * 16 + ln) * LQ, 0, half), o[nt]);
      o[nt] = wmma16(load_a_frag(Ps + (qr + ln) * LQ, 32, half),
                     load_b_frag(Vts + (nt * 16 + ln) * LQ, 32, half), o[nt]);
    }
    __syncthreads();
  }

  if (tid < 64) scale_s[tid] = (l_r > 0.0f) ? (1.0f / l_r) : 0.0f;
  __syncthreads();

  // write ctx f16 in [B,S,D] (sequence-major for the final projection)
#pragma unroll
  for (int nt = 0; nt < 4; ++nt) {
    const int d = nt * 16 + ln;
#pragma unroll
    for (int vv = 0; vv < 8; ++vv) {
      const int row = qr + vv + 8 * half;
      const int sg = q0 + row;
      const float val = o[nt][vv] * scale_s[row];
      ctx[((size_t)(bi * SEQ + sg)) * D_MODEL + h * HEAD_DIM + d] = (_Float16)val;
    }
  }
}

// ---------------------------------------------------------------------------
extern "C" void kernel_launch(void* const* d_in, const int* in_sizes, int n_in,
                              void* d_out, int out_size, void* d_ws, size_t ws_size,
                              hipStream_t stream) {
  (void)in_sizes; (void)n_in; (void)out_size; (void)ws_size;
  const float* query = (const float*)d_in[0];
  const float* key_i = (const float*)d_in[1];
  const float* value = (const float*)d_in[2];
  const unsigned char* mask = (const unsigned char*)d_in[3];  // jnp bool -> 1B
  const float* Wq = (const float*)d_in[4];
  const float* bq = (const float*)d_in[5];
  const float* Wk = (const float*)d_in[6];
  const float* bk = (const float*)d_in[7];
  const float* Wv = (const float*)d_in[8];
  const float* bv = (const float*)d_in[9];
  const float* Wo = (const float*)d_in[10];
  const float* bo = (const float*)d_in[11];

  const int M = BATCH * SEQ;                 // 8192
  const size_t per = (size_t)M * D_MODEL;    // 8,388,608 elems (16 MB f16 each)
  _Float16* qws = (_Float16*)d_ws;           // [B,H,S,hd]
  _Float16* kws = qws + per;
  _Float16* vws = kws + per;
  _Float16* ctxws = vws + per;               // [B,S,D]; total ws = 64 MB

  dim3 gblk(256), ggrid(D_MODEL / 128, M / 64);
  gemm_wmma_kernel<float, true><<<ggrid, gblk, 0, stream>>>(
      query, Wq, bq, (void*)qws, M, D_MODEL, D_MODEL);
  gemm_wmma_kernel<float, true><<<ggrid, gblk, 0, stream>>>(
      key_i, Wk, bk, (void*)kws, M, D_MODEL, D_MODEL);
  gemm_wmma_kernel<float, true><<<ggrid, gblk, 0, stream>>>(
      value, Wv, bv, (void*)vws, M, D_MODEL, D_MODEL);

  dim3 ablk(128), agrid(SEQ / 64, BATCH * N_HEADS);
  flash_attn_kernel<<<agrid, ablk, 0, stream>>>(qws, kws, vws, mask, ctxws);

  gemm_wmma_kernel<_Float16, false><<<ggrid, gblk, 0, stream>>>(
      ctxws, Wo, bo, d_out, M, D_MODEL, D_MODEL);
}
